// NoiseCfC_22660247454048
// MI455X (gfx1250) — matrix operand order repaired and checked by reading, not compiled, not run on hardware
//
#include <hip/hip_runtime.h>
#include <stdint.h>

// Problem dims (fixed by reference)
#define B_DIM  1024
#define T_DIM  512
#define I_DIM  64
#define H_DIM  128
#define BU_DIM 128

typedef __bf16 bf16;
typedef __attribute__((ext_vector_type(16))) __bf16 v16bf;
typedef __attribute__((ext_vector_type(8)))  __bf16 v8bf;
typedef __attribute__((ext_vector_type(8)))  float   v8f;

// ---------------------------------------------------------------------------
// WMMA wrapper: D = A(16x32 bf16) * B(32x16 bf16) + C(16x16 f32)
// ---------------------------------------------------------------------------
__device__ __forceinline__ v8f wmma_bf16(v16bf a, v16bf b, v8f c) {
  return __builtin_amdgcn_wmma_f32_16x16x32_bf16(
      /*neg_a=*/false, a, /*neg_b=*/false, b,
      /*c_mod=*/(short)0, c, /*reuse_a=*/false, /*reuse_b=*/false);
}

// Branchless activations on native transcendentals (v_exp_f32 / v_rcp_f32).
// Saturates correctly at +/-inf; bf16-pipeline precision is ample.
__device__ __forceinline__ float fast_tanh(float v) {
  float e = __expf(2.0f * v);
  return 1.0f - 2.0f * __builtin_amdgcn_rcpf(e + 1.0f);
}
__device__ __forceinline__ float fast_sigmoid(float v) {
  return __builtin_amdgcn_rcpf(1.0f + __expf(-v));
}

// ---------------------------------------------------------------------------
// B-matrix fragment (32x16, bf16), per ISA 7.12.2:
//   lane L: column n = L%16; elems 0..7 = K kb..kb+7, elems 8..15 = K kb+16..kb+23
// where kb = kbase + 8*(L/16). W stored column-major: WT[n][k], K = row length.
// ---------------------------------------------------------------------------
template <int K>
__device__ __forceinline__ v16bf load_bfrag(const bf16* __restrict__ WT,
                                            int col16, int kbase, int lane) {
  int n  = col16 + (lane & 15);
  int kb = kbase + ((lane >> 4) << 3);
  const bf16* p = WT + n * K + kb;
  v8bf lo = *(const v8bf*)(p);
  v8bf hi = *(const v8bf*)(p + 16);
  v16bf r;
#pragma unroll
  for (int i = 0; i < 8; ++i) { r[i] = lo[i]; r[i + 8] = hi[i]; }
  return r;
}

// ---------------------------------------------------------------------------
// A-matrix fragment (16x32, bf16), per ISA 7.12.2:
//   lane L: row m = L%16; elems 0..7 = K kb..kb+7, elems 8..15 = K kb+16..kb+23
//   where kb = kbase + 8*(L/16).   M is row-major with row stride RS.
// ---------------------------------------------------------------------------
template <int RS>
__device__ __forceinline__ v16bf load_afrag(const bf16* __restrict__ M,
                                            int kbase, int lane) {
  int m  = lane & 15;
  int k0 = kbase + ((lane >> 4) << 3);
  const bf16* p = M + m * RS + k0;
  v8bf lo = *(const v8bf*)(p);
  v8bf hi = *(const v8bf*)(p + 16);
  v16bf r;
#pragma unroll
  for (int i = 0; i < 8; ++i) { r[i] = lo[i]; r[i + 8] = hi[i]; }
  return r;
}

// A-fragment straight from global f32 x with gap-mask applied + bf16 convert.
__device__ __forceinline__ v16bf load_xfrag(const float* __restrict__ x,
                                            const unsigned char* __restrict__ gap,
                                            int b0, int t, int kbase, int lane) {
  int m  = lane & 15;
  int k0 = kbase + ((lane >> 4) << 3);
  size_t row = (size_t)(b0 + m);
  const float* p = x + (row * T_DIM + (size_t)t) * I_DIM + k0;
  float s = gap[row * T_DIM + t] ? 0.0f : 1.0f;
  float4 u0 = *(const float4*)(p);
  float4 u1 = *(const float4*)(p + 4);
  float4 u2 = *(const float4*)(p + 16);
  float4 u3 = *(const float4*)(p + 20);
  v16bf r;
  r[0]  = (bf16)(u0.x * s); r[1]  = (bf16)(u0.y * s);
  r[2]  = (bf16)(u0.z * s); r[3]  = (bf16)(u0.w * s);
  r[4]  = (bf16)(u1.x * s); r[5]  = (bf16)(u1.y * s);
  r[6]  = (bf16)(u1.z * s); r[7]  = (bf16)(u1.w * s);
  r[8]  = (bf16)(u2.x * s); r[9]  = (bf16)(u2.y * s);
  r[10] = (bf16)(u2.z * s); r[11] = (bf16)(u2.w * s);
  r[12] = (bf16)(u3.x * s); r[13] = (bf16)(u3.y * s);
  r[14] = (bf16)(u3.z * s); r[15] = (bf16)(u3.w * s);
  return r;
}

// Two deterministic N(0,1) values from one counter (Box-Muller pair, native
// v_log/v_sin/v_cos/v_sqrt). JAX threefry cannot be bit-matched; this is
// statistically equivalent and deterministic per output index.
__device__ __forceinline__ void gauss_noise2(uint32_t idx, float* n0, float* n1) {
  uint32_t a = idx * 0x9E3779B1u + 0x85EBCA77u;
  a ^= a >> 16; a *= 0x7FEB352Du; a ^= a >> 15; a *= 0x846CA68Bu; a ^= a >> 16;
  uint32_t b = (idx ^ 0xC2B2AE3Du) * 0x27D4EB2Fu;
  b ^= b >> 15; b *= 0x2C1B3C6Du; b ^= b >> 12; b *= 0x297A2D39u; b ^= b >> 15;
  float u1 = (float)(a >> 8) * 0x1p-24f + 0x1p-25f;   // (0,1)
  float u2 = (float)(b >> 8) * 0x1p-24f;
  float r  = sqrtf(-2.0f * __logf(u1));
  float th = 6.2831853071795864f * u2;
  *n0 = r * __cosf(th);
  *n1 = r * __sinf(th);
}

// ---------------------------------------------------------------------------
// Prep kernel: transpose weights to column-major bf16, fold Wt=Wta+Wtb,
// pack biases (bb | bf1 | bf2 | bta+btb) as f32.
// ---------------------------------------------------------------------------
__global__ void cfc_prep(const float* __restrict__ Wb,  const float* __restrict__ bb,
                         const float* __restrict__ Wf1, const float* __restrict__ bf1,
                         const float* __restrict__ Wf2, const float* __restrict__ bf2,
                         const float* __restrict__ Wta, const float* __restrict__ bta,
                         const float* __restrict__ Wtb, const float* __restrict__ btb,
                         bf16* __restrict__ wbxT, bf16* __restrict__ wbhT,
                         bf16* __restrict__ wf1T, bf16* __restrict__ wf2T,
                         bf16* __restrict__ wtT,  float* __restrict__ biases) {
  int i0 = blockIdx.x * blockDim.x + threadIdx.x;
  int nthr = gridDim.x * blockDim.x;
  for (int i = i0; i < 128 * 64; i += nthr) {         // Wbx^T [n][k], k<64
    int n = i >> 6, k = i & 63;
    wbxT[i] = (bf16)Wb[k * 128 + n];
  }
  for (int i = i0; i < 128 * 128; i += nthr) {        // [n][k] transposes
    int n = i >> 7, k = i & 127;
    wbhT[i] = (bf16)Wb[(64 + k) * 128 + n];
    wf1T[i] = (bf16)Wf1[k * 128 + n];
    wf2T[i] = (bf16)Wf2[k * 128 + n];
    wtT[i]  = (bf16)(Wta[k * 128 + n] + Wtb[k * 128 + n]);
  }
  for (int i = i0; i < 128; i += nthr) {
    biases[i]       = bb[i];
    biases[128 + i] = bf1[i];
    biases[256 + i] = bf2[i];
    biases[384 + i] = bta[i] + btb[i];
  }
}

// ---------------------------------------------------------------------------
// Recurrent kernel. 1 workgroup (4 waves) per 16-row batch tile; each wave
// owns a 32-column output chunk. Head weights + biases + h/f tiles in LDS;
// Wbx/Wbh B-fragments register-resident. The bias + x_t@Wbx partial sum is
// software-pipelined into the previous iteration's tail, so each loop head's
// critical path is only the 8 h@Wbh WMMAs. LDS occupancy gives 3 WG/WGP, so
// residual LDS waits are covered by wave interleave.
// ---------------------------------------------------------------------------
__global__ __launch_bounds__(128, 1)
void cfc_recurrent(const float* __restrict__ x, const float* __restrict__ hx,
                   const unsigned char* __restrict__ gap,
                   const bf16* __restrict__ wbxT, const bf16* __restrict__ wbhT,
                   const char* __restrict__ head_blob,   // wf1T|wf2T|wtT|biases
                   const float* __restrict__ nsc,
                   float* __restrict__ out) {
  extern __shared__ char smem[];
  bf16*  wf1s = (bf16*)(smem);                 // 32768 B
  bf16*  wf2s = (bf16*)(smem + 32768);         // 32768 B
  bf16*  wts  = (bf16*)(smem + 65536);         // 32768 B
  float* bias = (float*)(smem + 98304);        //  2048 B (bb|bf1|bf2|bt)
  bf16*  hbuf = (bf16*)(smem + 100352);        //  4096 B  16x128
  bf16*  fbuf = (bf16*)(smem + 104448);        //  4096 B  16x128

  const int tid   = threadIdx.x;
  const int lane  = tid & 31;
  const int col16 = (tid >> 5) * 32;           // this wave's 32-col chunk
  const int b0    = blockIdx.x * 16;           // batch tile base row
  const int lanen = lane & 15;
  const int hi8   = (lane >> 4) << 3;          // C-layout: m = v + hi8

  // Stage head weights + biases into LDS (one contiguous 100352B blob).
  {
    const int4* src = (const int4*)head_blob;
    int4* dst = (int4*)smem;
    for (int i = tid; i < 100352 / 16; i += 128) dst[i] = src[i];
  }
  // Init h tile from hx.
  for (int i = tid; i < 16 * 128; i += 128) {
    int m = i >> 7, c = i & 127;
    hbuf[i] = (bf16)hx[(size_t)(b0 + m) * H_DIM + c];
  }

  // Register-resident B-fragments for the f-GEMM (x:K=64, h:K=128).
  v16bf wbx[2][2], wbh[4][2];
#pragma unroll
  for (int kk = 0; kk < 2; ++kk)
#pragma unroll
    for (int nt = 0; nt < 2; ++nt)
      wbx[kk][nt] = load_bfrag<64>(wbxT, col16 + nt * 16, kk * 32, lane);
#pragma unroll
  for (int kk = 0; kk < 4; ++kk)
#pragma unroll
    for (int nt = 0; nt < 2; ++nt)
      wbh[kk][nt] = load_bfrag<128>(wbhT, col16 + nt * 16, kk * 32, lane);

  const float ns = *nsc;
  float* out_last = out + (size_t)B_DIM * T_DIM * H_DIM;

  // Pipelined partial sum: xacc = bb + x_t @ Wbx (recurrence-independent).
  v8f xacc[2];
  {
    float bv0 = bias[col16 + lanen], bv1 = bias[col16 + 16 + lanen];
#pragma unroll
    for (int v = 0; v < 8; ++v) { xacc[0][v] = bv0; xacc[1][v] = bv1; }
#pragma unroll
    for (int kk = 0; kk < 2; ++kk) {
      v16bf ax = load_xfrag(x, gap, b0, 0, kk * 32, lane);
#pragma unroll
      for (int nt = 0; nt < 2; ++nt) xacc[nt] = wmma_bf16(ax, wbx[kk][nt], xacc[nt]);
    }
  }

  __syncthreads();

  for (int t = 0; t < T_DIM; ++t) {
    // ---- f = lecun_tanh(xacc + h@Wbh): critical path is just these 8 WMMAs
    v8f facc[2];
#pragma unroll
    for (int nt = 0; nt < 2; ++nt) facc[nt] = xacc[nt];
#pragma unroll
    for (int kk = 0; kk < 4; ++kk) {
      v16bf ah = load_afrag<128>(hbuf, kk * 32, lane);
#pragma unroll
      for (int nt = 0; nt < 2; ++nt) facc[nt] = wmma_bf16(ah, wbh[kk][nt], facc[nt]);
    }

#pragma unroll
    for (int nt = 0; nt < 2; ++nt) {
      int c = col16 + nt * 16 + lanen;
#pragma unroll
      for (int v = 0; v < 8; ++v) {
        float fv = 1.7159f * fast_tanh(0.666f * facc[nt][v]);
        fbuf[(v + hi8) * 128 + c] = (bf16)fv;
      }
    }
    __syncthreads();   // f complete; everyone done reading hbuf

    // ---- heads: ff1, ff2, t_interp over full f (K=128) ----
    v8f a1[2], a2[2], a3[2];
#pragma unroll
    for (int nt = 0; nt < 2; ++nt) {
      int c = col16 + nt * 16 + lanen;
      float c1 = bias[128 + c], c2 = bias[256 + c], c3 = bias[384 + c];
#pragma unroll
      for (int v = 0; v < 8; ++v) { a1[nt][v] = c1; a2[nt][v] = c2; a3[nt][v] = c3; }
    }
#pragma unroll
    for (int kk = 0; kk < 4; ++kk) {
      v16bf fa = load_afrag<128>(fbuf, kk * 32, lane);
#pragma unroll
      for (int nt = 0; nt < 2; ++nt) {
        a1[nt] = wmma_bf16(fa, load_bfrag<128>(wf1s, col16 + nt * 16, kk * 32, lane), a1[nt]);
        a2[nt] = wmma_bf16(fa, load_bfrag<128>(wf2s, col16 + nt * 16, kk * 32, lane), a2[nt]);
        a3[nt] = wmma_bf16(fa, load_bfrag<128>(wts,  col16 + nt * 16, kk * 32, lane), a3[nt]);
      }
    }

    // Pipeline: bias + x_{t+1}@Wbx now, hidden behind head WMMAs in flight.
    if (t + 1 < T_DIM) {
      float bv0 = bias[col16 + lanen], bv1 = bias[col16 + 16 + lanen];
#pragma unroll
      for (int v = 0; v < 8; ++v) { xacc[0][v] = bv0; xacc[1][v] = bv1; }
#pragma unroll
      for (int kk = 0; kk < 2; ++kk) {
        v16bf ax = load_xfrag(x, gap, b0, t + 1, kk * 32, lane);
#pragma unroll
        for (int nt = 0; nt < 2; ++nt) xacc[nt] = wmma_bf16(ax, wbx[kk][nt], xacc[nt]);
      }
    }

    // ---- elementwise gate + h update + noisy output (noise in BM pairs) ----
#pragma unroll
    for (int nt = 0; nt < 2; ++nt) {
      int c = col16 + nt * 16 + lanen;
#pragma unroll
      for (int v = 0; v < 8; v += 2) {
        float hn[2];
#pragma unroll
        for (int u = 0; u < 2; ++u) {
          float f1 = fast_tanh(a1[nt][v + u]);
          float f2 = fast_tanh(a2[nt][v + u]);
          float ti = fast_sigmoid(a3[nt][v + u]);
          hn[u] = f1 + ti * (f2 - f1);
          hbuf[(v + u + hi8) * 128 + c] = (bf16)hn[u];
        }
        int m0 = v + hi8;
        size_t oi0 = ((size_t)(b0 + m0) * T_DIM + (size_t)t) * H_DIM + c;
        size_t oi1 = oi0 + (size_t)T_DIM * H_DIM;   // row m0+1
        float g0, g1;
        gauss_noise2((uint32_t)oi0, &g0, &g1);
        float ov0 = hn[0] + ns * g0;
        float ov1 = hn[1] + ns * g1;
        out[oi0] = ov0;
        out[oi1] = ov1;
        if (t == T_DIM - 1) {
          out_last[(size_t)(b0 + m0) * H_DIM + c]     = ov0;
          out_last[(size_t)(b0 + m0 + 1) * H_DIM + c] = ov1;
        }
      }
    }
    __syncthreads();       // h published for next step
  }
}

// ---------------------------------------------------------------------------
extern "C" void kernel_launch(void* const* d_in, const int* in_sizes, int n_in,
                              void* d_out, int out_size, void* d_ws, size_t ws_size,
                              hipStream_t stream) {
  const float*         x    = (const float*)d_in[0];
  const float*         hx   = (const float*)d_in[1];
  const unsigned char* gap  = (const unsigned char*)d_in[2];
  const float* Wb  = (const float*)d_in[3];  const float* bb  = (const float*)d_in[4];
  const float* Wf1 = (const float*)d_in[5];  const float* bf1 = (const float*)d_in[6];
  const float* Wf2 = (const float*)d_in[7];  const float* bf2 = (const float*)d_in[8];
  const float* Wta = (const float*)d_in[9];  const float* bta = (const float*)d_in[10];
  const float* Wtb = (const float*)d_in[11]; const float* btb = (const float*)d_in[12];
  const float* nsc = (const float*)d_in[13];
  float* out = (float*)d_out;

  // Workspace layout (bytes):
  //   [0,16384)        WbxT  bf16 [128n][64k]
  //   [16384,49152)    WbhT  bf16 [128n][128k]
  //   [49152,81920)    Wf1T  bf16 [128n][128k]   \
  //   [81920,114688)   Wf2T  bf16 [128n][128k]    } contiguous "head blob"
  //   [114688,147456)  WtT   bf16 [128n][128k]    } + biases => 100352 B
  //   [147456,149504)  biases f32 [4][128]       /
  char* ws = (char*)d_ws;
  bf16*  wbxT   = (bf16*)(ws);
  bf16*  wbhT   = (bf16*)(ws + 16384);
  bf16*  wf1T   = (bf16*)(ws + 49152);
  bf16*  wf2T   = (bf16*)(ws + 81920);
  bf16*  wtT    = (bf16*)(ws + 114688);
  float* biases = (float*)(ws + 147456);

  cfc_prep<<<32, 256, 0, stream>>>(Wb, bb, Wf1, bf1, Wf2, bf2, Wta, bta, Wtb, btb,
                                   wbxT, wbhT, wf1T, wf2T, wtT, biases);

  const size_t shmem = 108544;  // 96K head weights + 2K biases + 8K h/f tiles
  (void)hipFuncSetAttribute((const void*)cfc_recurrent,
                            hipFuncAttributeMaxDynamicSharedMemorySize, (int)shmem);
  cfc_recurrent<<<B_DIM / 16, 128, shmem, stream>>>(x, hx, gap, wbxT, wbhT,
                                                    ws + 49152, nsc, out);
}
